// SequenceAttentionMM_12163347383239
// MI455X (gfx1250) — compile-verified, matrix operand order checked by reading, not run on hardware
//
#include <hip/hip_runtime.h>
#include <hip/hip_bf16.h>

namespace {

constexpr int kB  = 16;
constexpr int kLU = 2048;
constexpr int kLV = 2048;
constexpr int kD  = 1024;
constexpr int kBN = 128;   // v rows per flash-attention iteration

typedef __attribute__((ext_vector_type(16))) __bf16 v16b;
typedef __attribute__((ext_vector_type(8)))  float  v8f;

union FragU { int4 q[2]; v16b v; };

// Load a 16x32 bf16 WMMA operand fragment (A layout; B mirrored).
// ISA layout: lanes 0-15 hold row (lane&15) K={0..7,16..23};
// lanes 16-31 hold the same row K={8..15,24..31}. Two 16B chunks per lane.
__device__ __forceinline__ v16b load_frag(const __bf16* base, int ld, int lane) {
  const int r = lane & 15;
  const int k = (lane & 16) ? 8 : 0;
  const __bf16* p = base + (size_t)r * ld + k;
  FragU u;
  u.q[0] = *(const int4*)(p);
  u.q[1] = *(const int4*)(p + 16);
  return u.v;
}

__device__ __forceinline__ v8f wmma_bf16(v16b a, v16b b, v8f c) {
  return __builtin_amdgcn_wmma_f32_16x16x32_bf16(false, a, false, b, (short)0, c,
                                                 false, false);
}

__device__ __forceinline__ v8f zero_v8f() {
  v8f z = {0.f, 0.f, 0.f, 0.f, 0.f, 0.f, 0.f, 0.f};
  return z;
}

// ---------------------------------------------------------------------------
// Wb = bf16(W), 4 elements per thread, packed 8B stores.
// ---------------------------------------------------------------------------
__global__ void cast_w_kernel(const float* __restrict__ W, __bf16* __restrict__ Wb) {
  const int i = (blockIdx.x * blockDim.x + threadIdx.x) * 4;
  const float4 a = *(const float4*)(W + i);
  union { __bf16 h[4]; short4 s; } pk;
  pk.h[0] = (__bf16)a.x; pk.h[1] = (__bf16)a.y;
  pk.h[2] = (__bf16)a.z; pk.h[3] = (__bf16)a.w;
  *(short4*)(Wb + i) = pk.s;
}

// ---------------------------------------------------------------------------
// vt[b][d][lv] = bf16(v[b][lv][d])  — K-contiguous B-operand for the PV GEMM
// ---------------------------------------------------------------------------
__global__ void transpose_v_kernel(const float* __restrict__ v, __bf16* __restrict__ vt) {
  __shared__ float tile[32][33];
  const int b   = blockIdx.z;
  const int lv0 = blockIdx.x * 32;
  const int d0  = blockIdx.y * 32;
  const int t   = threadIdx.x;
  const int r   = t >> 3;        // 0..31
  const int c4  = (t & 7) * 4;   // 0,4,...,28

  const float4 src = *(const float4*)(v + ((size_t)b * kLV + lv0 + r) * kD + d0 + c4);
  tile[r][c4 + 0] = src.x; tile[r][c4 + 1] = src.y;
  tile[r][c4 + 2] = src.z; tile[r][c4 + 3] = src.w;
  __syncthreads();

  union { __bf16 h[4]; short4 s; } pk;
  #pragma unroll
  for (int i = 0; i < 4; ++i) pk.h[i] = (__bf16)tile[c4 + i][r];
  *(short4*)(vt + ((size_t)b * kD + d0 + r) * kLV + lv0 + c4) = pk.s;
}

// ---------------------------------------------------------------------------
// Y = relu(X @ W^T + bias) in bf16.   X: f32 [Nrows, D], Wb: bf16 [D, D].
// WG tile 64(M) x 128(N); 8 waves = 4(M) x 2(N); each wave 16x64 output.
// Double-buffered LDS staging of the f32->bf16 A tile (one barrier / K-step).
// ---------------------------------------------------------------------------
__global__ __launch_bounds__(256, 2)
void proj_relu_kernel(const float* __restrict__ X, const __bf16* __restrict__ Wb,
                      const float* __restrict__ bias, __bf16* __restrict__ Y) {
  __shared__ __bf16 Ash[2][64 * 40];         // 2 x (64x32, padded ld=40, 16B-aligned rows)
  const int t    = threadIdx.x;
  const int lane = t & 31;
  const int w    = t >> 5;
  const int wm   = w & 3;                    // M sub-tile (16 rows)
  const int wn   = w >> 2;                   // N half (64 cols)
  const int l16  = lane & 15;
  const int hs   = (lane >> 4) & 1;

  const int m0 = blockIdx.x * 64;
  const int n0 = blockIdx.y * 128;

  v8f C[4];
  #pragma unroll
  for (int i = 0; i < 4; ++i) C[i] = zero_v8f();

  const int srow = t >> 2;                   // 0..63
  const int scol = (t & 3) * 8;              // 0,8,16,24

  auto stage = [&](int k0, int buf) {
    const float* src = X + (size_t)(m0 + srow) * kD + k0 + scol;
    const float4 a  = *(const float4*)(src);
    const float4 b2 = *(const float4*)(src + 4);
    union { __bf16 h[8]; int4 q; } cv;
    cv.h[0] = (__bf16)a.x;  cv.h[1] = (__bf16)a.y;
    cv.h[2] = (__bf16)a.z;  cv.h[3] = (__bf16)a.w;
    cv.h[4] = (__bf16)b2.x; cv.h[5] = (__bf16)b2.y;
    cv.h[6] = (__bf16)b2.z; cv.h[7] = (__bf16)b2.w;
    *(int4*)(&Ash[buf][srow * 40 + scol]) = cv.q;
  };

  stage(0, 0);
  __syncthreads();

  int cur = 0;
  for (int k0 = 0; k0 < kD; k0 += 32) {
    if (k0 + 32 < kD) stage(k0 + 32, cur ^ 1);   // overlap next stage with compute

    const v16b Af = load_frag(&Ash[cur][wm * 16 * 40], 40, lane);
    v16b Bf[4];
    #pragma unroll
    for (int nt = 0; nt < 4; ++nt)
      Bf[nt] = load_frag(Wb + (size_t)(n0 + wn * 64 + nt * 16) * kD + k0, kD, lane);
    #pragma unroll
    for (int nt = 0; nt < 4; ++nt)
      C[nt] = wmma_bf16(Af, Bf[nt], C[nt]);

    __syncthreads();
    cur ^= 1;
  }

  #pragma unroll
  for (int nt = 0; nt < 4; ++nt) {
    const int n  = n0 + wn * 64 + nt * 16 + l16;
    const float bn = bias[n];
    #pragma unroll
    for (int j = 0; j < 8; ++j) {
      const int m = m0 + wm * 16 + j + 8 * hs;   // C layout: vgpr j -> row j (+8 hi half)
      float y = C[nt][j] + bn;
      y = fmaxf(y, 0.f);
      Y[(size_t)m * kD + n] = (__bf16)y;
    }
  }
}

// ---------------------------------------------------------------------------
// Flash attention: one 16-row u tile per 256-thread workgroup (8 waves).
// Each wave owns a 128-wide D slice of O. Per 128-row v block:
//   wave w computes S tile for v rows [v0+16w, +16) over full K=1024,
//   block-wide online softmax through LDS, then O += P @ V from vt.
// B-fragment streams are software-pipelined two loads deep.
// ---------------------------------------------------------------------------
__global__ __launch_bounds__(256, 1)
void attn_kernel(const __bf16* __restrict__ ubp, const __bf16* __restrict__ vbp,
                 const __bf16* __restrict__ vt, const unsigned char* __restrict__ vmask,
                 float* __restrict__ out) {
  __shared__ float  S_lds[16][kBN];          // 8 KB
  __shared__ __bf16 P_lds[16][kBN];          // 4 KB
  __shared__ float  m_s[16], l_s[16], scl_s[16];

  const int b    = blockIdx.y;
  const int u0   = blockIdx.x * 16;
  const int t    = threadIdx.x;
  const int lane = t & 31;
  const int w    = t >> 5;
  const int l16  = lane & 15;
  const int hs   = (lane >> 4) & 1;

  // Q: 16 rows x 1024 K, held in registers as 32 fragments (256 VGPRs)
  const __bf16* Qbase = ubp + ((size_t)b * kLU + u0) * kD;
  v16b Qf[32];
  #pragma unroll
  for (int kc = 0; kc < 32; ++kc) Qf[kc] = load_frag(Qbase + kc * 32, kD, lane);

  v8f O[8];
  #pragma unroll
  for (int i = 0; i < 8; ++i) O[i] = zero_v8f();

  if (t < 16) { m_s[t] = -__builtin_inff(); l_s[t] = 0.f; }
  __syncthreads();

  for (int v0 = 0; v0 < kLV; v0 += kBN) {
    const __bf16* Kbase = vbp + ((size_t)b * kLV + v0 + w * 16) * kD;

    // prefetch next block's K rows toward cache while we compute this block
    if (v0 + kBN < kLV) {
      const __bf16* Kn = Kbase + (size_t)kBN * kD;
      __builtin_prefetch(Kn + (size_t)l16 * kD + hs * 256, 0, 0);
      __builtin_prefetch(Kn + (size_t)l16 * kD + hs * 256 + 512, 0, 0);
    }

    // ---- S = Q * K^T for this wave's 16 v rows (f32 acc), 2-deep pipeline ----
    v8f S = zero_v8f();
    {
      v16b f0 = load_frag(Kbase, kD, lane);
      v16b f1 = load_frag(Kbase + 32, kD, lane);
      #pragma unroll
      for (int kc = 0; kc < 32; ++kc) {
        const v16b curf = f0;
        f0 = f1;
        if (kc + 2 < 32) f1 = load_frag(Kbase + (kc + 2) * 32, kD, lane);
        S = wmma_bf16(Qf[kc], curf, S);
      }
    }

    // padding mask: lane's 8 acc elements all share one v column
    const int  vrow = v0 + w * 16 + l16;
    const bool msk  = vmask[(size_t)b * kLV + vrow] != 0;
    #pragma unroll
    for (int j = 0; j < 8; ++j)
      S_lds[j + 8 * hs][w * 16 + l16] = msk ? -__builtin_inff() : S[j];
    __syncthreads();

    // ---- online softmax over the 16 x 128 block (16 threads per row) ----
    {
      const int row = t >> 4;
      const int sub = t & 15;
      float vals[8];
      float mx = -__builtin_inff();
      #pragma unroll
      for (int c = 0; c < 8; ++c) { vals[c] = S_lds[row][sub * 8 + c]; mx = fmaxf(mx, vals[c]); }
      #pragma unroll
      for (int off = 8; off >= 1; off >>= 1) mx = fmaxf(mx, __shfl_xor(mx, off, 16));
      const float m_old = m_s[row];
      const float m_new = fmaxf(m_old, mx);
      float sum = 0.f;
      #pragma unroll
      for (int c = 0; c < 8; ++c) {
        const float p = __expf(vals[c] - m_new);
        sum += p;
        P_lds[row][sub * 8 + c] = (__bf16)p;
      }
      #pragma unroll
      for (int off = 8; off >= 1; off >>= 1) sum += __shfl_xor(sum, off, 16);
      if (sub == 0) {
        const float corr = __expf(m_old - m_new);
        scl_s[row] = corr;
        l_s[row]   = l_s[row] * corr + sum;
        m_s[row]   = m_new;
      }
    }
    __syncthreads();

    // ---- rescale O, then O += P @ V over this wave's 128 D columns ----
    float corr[8];
    #pragma unroll
    for (int j = 0; j < 8; ++j) corr[j] = scl_s[j + 8 * hs];
    #pragma unroll
    for (int nt = 0; nt < 8; ++nt) {
      #pragma unroll
      for (int j = 0; j < 8; ++j) O[nt][j] *= corr[j];
    }

    v16b Pf[4];
    #pragma unroll
    for (int kc = 0; kc < 4; ++kc) Pf[kc] = load_frag(&P_lds[0][0] + kc * 32, kBN, lane);

    const __bf16* Vb = vt + ((size_t)b * kD + w * 128) * (size_t)kLV + v0;
    auto vaddr = [&](int i) {
      return Vb + (size_t)((i >> 2) * 16) * kLV + (i & 3) * 32;
    };
    {
      v16b f0 = load_frag(vaddr(0), kLV, lane);
      v16b f1 = load_frag(vaddr(1), kLV, lane);
      #pragma unroll
      for (int i = 0; i < 32; ++i) {
        const v16b curf = f0;
        f0 = f1;
        if (i + 2 < 32) f1 = load_frag(vaddr(i + 2), kLV, lane);
        O[i >> 2] = wmma_bf16(Pf[i & 3], curf, O[i >> 2]);
      }
    }
    __syncthreads();   // protect S_lds/P_lds for next block
  }

  // ---- epilogue: divide by softmax denominator, store f32 ----
  float linv[8];
  #pragma unroll
  for (int j = 0; j < 8; ++j) linv[j] = 1.f / l_s[j + 8 * hs];
  float* Ob = out + ((size_t)b * kLU + u0) * kD + w * 128;
  #pragma unroll
  for (int nt = 0; nt < 8; ++nt) {
    #pragma unroll
    for (int j = 0; j < 8; ++j)
      Ob[(size_t)(j + 8 * hs) * kD + nt * 16 + l16] = O[nt][j] * linv[j];
  }
}

}  // namespace

extern "C" void kernel_launch(void* const* d_in, const int* in_sizes, int n_in,
                              void* d_out, int out_size, void* d_ws, size_t ws_size,
                              hipStream_t stream) {
  (void)in_sizes; (void)n_in; (void)out_size; (void)ws_size;

  const float*         u     = (const float*)d_in[0];          // [B, LU, D] f32
  const float*         v     = (const float*)d_in[1];          // [B, LV, D] f32
  const unsigned char* vmask = (const unsigned char*)d_in[2];  // [B, LV] bool
  const float*         W     = (const float*)d_in[3];          // [D, D] f32
  const float*         bias  = (const float*)d_in[4];          // [D] f32
  float*               out   = (float*)d_out;                  // [B, LU, D] f32

  // workspace layout (~194 MB): Wb | ub_ | vb_ | vt
  char* ws = (char*)d_ws;
  __bf16* Wb  = (__bf16*)ws;  ws += (size_t)kD * kD * sizeof(__bf16);
  __bf16* ubp = (__bf16*)ws;  ws += (size_t)kB * kLU * kD * sizeof(__bf16);
  __bf16* vbp = (__bf16*)ws;  ws += (size_t)kB * kLV * kD * sizeof(__bf16);
  __bf16* vt  = (__bf16*)ws;

  cast_w_kernel<<<(kD * kD) / 1024, 256, 0, stream>>>(W, Wb);
  transpose_v_kernel<<<dim3(kLV / 32, kD / 32, kB), 256, 0, stream>>>(v, vt);
  proj_relu_kernel<<<dim3((kB * kLU) / 64, kD / 128), 256, 0, stream>>>(u, Wb, bias, ubp);
  proj_relu_kernel<<<dim3((kB * kLV) / 64, kD / 128), 256, 0, stream>>>(v, Wb, bias, vbp);
  attn_kernel<<<dim3(kLU / 16, kB), 256, 0, stream>>>(ubp, vbp, vt, vmask, out);
}
